// RNNModel_3728031613596
// MI455X (gfx1250) — matrix operand order repaired and checked by reading, not compiled, not run on hardware
//
#include <hip/hip_runtime.h>
#include <stdint.h>

typedef __attribute__((ext_vector_type(16))) __bf16 v16bf;
typedef __attribute__((ext_vector_type(8)))  float  v8f;

#define T_SEQ 1024
#define BATCH 128
#define EDIM  312
#define H1    256
#define H2    128
#define G1N   768   // 3*H1
#define G2N   384   // 3*H2

// ---------- helpers ----------

__device__ __forceinline__ unsigned short f2bf(float f) {
  unsigned u = __float_as_uint(f);
  unsigned r = u + 0x7FFFu + ((u >> 16) & 1u);   // RNE
  return (unsigned short)(r >> 16);
}
__device__ __forceinline__ unsigned pk2(float a, float b) {
  return ((unsigned)f2bf(b) << 16) | (unsigned)f2bf(a);
}

// fast activations: v_exp_f32 + v_rcp_f32, no IEEE divide expansion
__device__ __forceinline__ float sigmoidf_(float x) {
  return __builtin_amdgcn_rcpf(1.0f + __expf(-x));
}
__device__ __forceinline__ float tanhf_(float x) {
  // 1 - 2/(e^{2x}+1); exact limits at +/-inf
  return 1.0f - 2.0f * __builtin_amdgcn_rcpf(__expf(2.0f * x) + 1.0f);
}

union Frag { unsigned int u[8]; v16bf v; };

// A-fragment (16x32 bf16) from LDS, row-major bf16 with row stride Kdim.
// m = lane&15, half = lane>>4; vgpr v: kbase = (v>>2)*16 + half*8 + (v&3)*2
// (per-lane bytes are two contiguous 16B runs -> 2x ds_load_b128)
__device__ __forceinline__ void load_afrag(Frag& a, const unsigned short* hbf,
                                           int Kdim, int kt, int lane) {
  const int m = lane & 15, half = lane >> 4;
  const unsigned short* row = hbf + m * Kdim + kt * 32;
#pragma unroll
  for (int v = 0; v < 8; ++v) {
    int kbase = ((v >> 2) << 4) + (half << 3) + ((v & 3) << 1);
    a.u[v] = *(const unsigned int*)(row + kbase);
  }
}

// B-fragment: pre-swizzled, each lane reads 32 contiguous bytes (2x global_load_b128).
__device__ __forceinline__ void load_bfrag(Frag& b, const unsigned int* wsw,
                                           int tile, int lane) {
  const uint4* p = (const uint4*)(wsw + (size_t)tile * 256 + lane * 8);
  uint4 x0 = p[0], x1 = p[1];
  b.u[0] = x0.x; b.u[1] = x0.y; b.u[2] = x0.z; b.u[3] = x0.w;
  b.u[4] = x1.x; b.u[5] = x1.y; b.u[6] = x1.z; b.u[7] = x1.w;
}

#define WMMA_BF16(A, B, C) \
  __builtin_amdgcn_wmma_f32_16x16x32_bf16(false, (A).v, false, (B).v, (short)0, (C), false, false)

// ---------- K0a: G1 = emb @ w_ih1^T + b_ih1  (128 x 768, tiny) ----------
__global__ void k_gemb(const float* __restrict__ emb, const float* __restrict__ w_ih1,
                       const float* __restrict__ b_ih1, float* __restrict__ G1) {
  int idx = blockIdx.x * blockDim.x + threadIdx.x;   // exactly 128*768 threads
  int v = idx / G1N, g = idx % G1N;
  float s = b_ih1[g];
  const float* er = emb + v * EDIM;
  const float* wr = w_ih1 + g * EDIM;
  for (int e = 0; e < EDIM; ++e) s += er[e] * wr[e];
  G1[idx] = s;
}

// ---------- K0b: swizzle weight W(GxK f32, row-major) into bf16 B-fragment tiles
__global__ void k_swz(const float* __restrict__ W, unsigned int* __restrict__ out,
                      int G, int K) {
  int idx = blockIdx.x * blockDim.x + threadIdx.x;   // exactly (G/16)*(K/32)*256
  int tile = idx >> 8, r = idx & 255;
  int l = r >> 3, v = r & 7;
  int KT = K >> 5;
  int nt = tile / KT, kt = tile - nt * KT;
  int n = nt * 16 + (l & 15);
  int half = l >> 4;
  int k0 = kt * 32 + ((v >> 2) << 4) + (half << 3) + ((v & 3) << 1);
  unsigned short lo = f2bf(W[n * K + k0]);
  unsigned short hi = f2bf(W[n * K + k0 + 1]);
  out[idx] = ((unsigned)hi << 16) | (unsigned)lo;
}

// ---------- K1: layer-1 GRU recurrence + LayerNorm, one 16-batch slice per WG ----------
// 8 waves; wave w owns h-column tiles jt in {2w, 2w+1}; gate tiles {jt, 16+jt, 32+jt}.
// All 48 B-fragments (384 VGPRs) stay register-resident across the whole sequence.
__global__ void __launch_bounds__(256) k_layer1(
    const int* __restrict__ token, const float* __restrict__ G1,
    const unsigned int* __restrict__ whh_sw, const float* __restrict__ b_hh,
    const float* __restrict__ ln_g, const float* __restrict__ ln_b,
    unsigned short* __restrict__ hn_out) {
  __shared__ unsigned short h_bf[16 * H1];   // bf16 state for A-fragments
  __shared__ float          h_f32[16 * H1];  // f32 state for z*h blend
  __shared__ float          red_s[16 * 16];
  __shared__ float          red_q[16 * 16];
  __shared__ float          ln_lds[2][H1];   // staged LayerNorm params

  const int tid = threadIdx.x;
  const int lane = tid & 31;
  const int wave = tid >> 5;
  const int half = lane >> 4;
  const int ncol = lane & 15;
  const int bbase = blockIdx.x * 16;
  const int jt0 = wave * 2;

  for (int i = tid; i < 16 * H1; i += 256) { h_f32[i] = 0.0f; h_bf[i] = 0; }
  for (int i = tid; i < H1; i += 256) { ln_lds[0][i] = ln_g[i]; ln_lds[1][i] = ln_b[i]; }

  float bh_r[2], bh_z[2], bh_n[2];
#pragma unroll
  for (int q = 0; q < 2; ++q) {
    int j = (jt0 + q) * 16 + ncol;
    bh_r[q] = b_hh[j];
    bh_z[q] = b_hh[H1 + j];
    bh_n[q] = b_hh[2 * H1 + j];
  }

  // persistent register-resident recurrent weights: B[q][gate][kt]
  Frag B[2][3][8];
#pragma unroll
  for (int q = 0; q < 2; ++q) {
    int jt = jt0 + q;
#pragma unroll
    for (int kt = 0; kt < 8; ++kt) {
      load_bfrag(B[q][0][kt], whh_sw, (jt)      * 8 + kt, lane);
      load_bfrag(B[q][1][kt], whh_sw, (16 + jt) * 8 + kt, lane);
      load_bfrag(B[q][2][kt], whh_sw, (32 + jt) * 8 + kt, lane);
    }
  }
  __syncthreads();

#pragma unroll 1
  for (int t = 0; t < T_SEQ; ++t) {
    // issue token gathers early so their latency hides under the WMMA chain
    int tokv[8];
#pragma unroll
    for (int r = 0; r < 8; ++r)
      tokv[r] = token[(size_t)t * BATCH + bbase + (half << 3) + r];

    v8f zero = {};
    v8f acc_r[2] = {zero, zero}, acc_z[2] = {zero, zero}, acc_n[2] = {zero, zero};

    // hg = h @ w_hh1^T  (M=16, K=256; per wave: 2 col-tiles x 3 gates)
#pragma unroll
    for (int kt = 0; kt < 8; ++kt) {
      Frag a; load_afrag(a, h_bf, H1, kt, lane);
#pragma unroll
      for (int q = 0; q < 2; ++q) {
        acc_r[q] = WMMA_BF16(a, B[q][0][kt], acc_r[q]);
        acc_z[q] = WMMA_BF16(a, B[q][1][kt], acc_z[q]);
        acc_n[q] = WMMA_BF16(a, B[q][2][kt], acc_n[q]);
      }
    }

    // gate math, wave-local
#pragma unroll
    for (int r = 0; r < 8; ++r) {
      int m = r + (half << 3);
      const float* grow = G1 + tokv[r] * G1N;
#pragma unroll
      for (int q = 0; q < 2; ++q) {
        int j = (jt0 + q) * 16 + ncol;
        float xr = grow[j], xz = grow[H1 + j], xn = grow[2 * H1 + j];
        float rg = sigmoidf_(xr + acc_r[q][r] + bh_r[q]);
        float zg = sigmoidf_(xz + acc_z[q][r] + bh_z[q]);
        float ng = tanhf_(xn + rg * (acc_n[q][r] + bh_n[q]));
        float hp = h_f32[m * H1 + j];
        float hnew = (1.0f - zg) * ng + zg * hp;
        h_f32[m * H1 + j] = hnew;
      }
    }
    __syncthreads();

    // LayerNorm (eps = 256, faithful) + write bf16 hn and refresh bf16 state
    {
      int m = tid >> 4, tx = tid & 15;
      float s = 0.0f, qq = 0.0f, hv[16];
#pragma unroll
      for (int c = 0; c < 16; ++c) {
        float h = h_f32[m * H1 + tx * 16 + c];
        hv[c] = h; s += h; qq += h * h;
      }
      red_s[m * 16 + tx] = s; red_q[m * 16 + tx] = qq;
      __syncthreads();
      float su = 0.0f, sq = 0.0f;
#pragma unroll
      for (int c = 0; c < 16; ++c) { su += red_s[m * 16 + c]; sq += red_q[m * 16 + c]; }
      float mu = su * (1.0f / H1);
      float var = sq * (1.0f / H1) - mu * mu;
      float rstd = rsqrtf(var + 256.0f);

      // pack 16 bf16 hn values (2x b128 global store) and 16 bf16 state values (2x b128 ds store)
      unsigned po[8], ph[8];
#pragma unroll
      for (int c = 0; c < 8; ++c) {
        int j0 = tx * 16 + 2 * c;
        float a0 = (hv[2 * c]     - mu) * rstd * ln_lds[0][j0]     + ln_lds[1][j0];
        float a1 = (hv[2 * c + 1] - mu) * rstd * ln_lds[0][j0 + 1] + ln_lds[1][j0 + 1];
        po[c] = pk2(a0, a1);
        ph[c] = pk2(hv[2 * c], hv[2 * c + 1]);
      }
      size_t orow = ((size_t)t * BATCH + bbase + m) * (size_t)H1;
      uint4* op = (uint4*)(hn_out + orow + tx * 16);
      uint4 o0; o0.x = po[0]; o0.y = po[1]; o0.z = po[2]; o0.w = po[3];
      uint4 o1; o1.x = po[4]; o1.y = po[5]; o1.z = po[6]; o1.w = po[7];
      op[0] = o0; op[1] = o1;
      uint4* hp4 = (uint4*)(h_bf + m * H1 + tx * 16);
      uint4 h0; h0.x = ph[0]; h0.y = ph[1]; h0.z = ph[2]; h0.w = ph[3];
      uint4 h1; h1.x = ph[4]; h1.y = ph[5]; h1.z = ph[6]; h1.w = ph[7];
      hp4[0] = h0; hp4[1] = h1;
    }
    __syncthreads();
  }
}

// ---------- K2: layer-2 bidirectional GRU; blockIdx.x: [0,8)=fwd slices, [8,16)=bwd ----------
// 8 waves; wave w owns col tile jt=w; gate tiles {w, 8+w, 16+w}.
// All 36 B-fragments (288 VGPRs) register-resident; hn staging is double-buffered so the
// 8KB global->LDS copy for step s+1 overlaps step s's gate math.
__global__ void __launch_bounds__(256) k_layer2(
    const unsigned short* __restrict__ hn_ws,
    const unsigned int* __restrict__ wih_f, const unsigned int* __restrict__ whh_f,
    const float* __restrict__ bih_f, const float* __restrict__ bhh_f,
    const unsigned int* __restrict__ wih_b, const unsigned int* __restrict__ whh_b,
    const float* __restrict__ bih_b, const float* __restrict__ bhh_b,
    float* __restrict__ out) {
  __shared__ unsigned short hn_lds[2][16 * H1];  // double-buffered input slice
  __shared__ unsigned short h_bf[16 * H2];
  __shared__ float          h_f32[16 * H2];

  const int tid = threadIdx.x;
  const int lane = tid & 31;
  const int wave = tid >> 5;
  const int half = lane >> 4;
  const int ncol = lane & 15;
  const int dir = blockIdx.x >> 3;
  const int bbase = (blockIdx.x & 7) * 16;

  const unsigned int* wih = dir ? wih_b : wih_f;
  const unsigned int* whh = dir ? whh_b : whh_f;
  const float* bih = dir ? bih_b : bih_f;
  const float* bhh = dir ? bhh_b : bhh_f;

  for (int i = tid; i < 16 * H2; i += 256) { h_f32[i] = 0.0f; h_bf[i] = 0; }

  const int jt = wave;
  const int j = jt * 16 + ncol;
  const float bi_r = bih[j], bi_z = bih[H2 + j], bi_n = bih[2 * H2 + j];
  const float bh_r = bhh[j], bh_z = bhh[H2 + j], bh_n = bhh[2 * H2 + j];

  // persistent register-resident weights
  Frag Bx[3][8];   // input weights: 3 gates x 8 kt
  Frag Bh[3][4];   // hidden weights: 3 gates x 4 kt
#pragma unroll
  for (int kt = 0; kt < 8; ++kt) {
    load_bfrag(Bx[0][kt], wih, (jt)      * 8 + kt, lane);
    load_bfrag(Bx[1][kt], wih, (8 + jt)  * 8 + kt, lane);
    load_bfrag(Bx[2][kt], wih, (16 + jt) * 8 + kt, lane);
  }
#pragma unroll
  for (int kt = 0; kt < 4; ++kt) {
    load_bfrag(Bh[0][kt], whh, (jt)      * 4 + kt, lane);
    load_bfrag(Bh[1][kt], whh, (8 + jt)  * 4 + kt, lane);
    load_bfrag(Bh[2][kt], whh, (16 + jt) * 4 + kt, lane);
  }

  // prologue: stage step 0
  {
    const int t0 = dir ? (T_SEQ - 1) : 0;
    const uint4* src = (const uint4*)(hn_ws + ((size_t)t0 * BATCH + bbase) * H1);
    uint4* dst = (uint4*)hn_lds[0];
    dst[tid] = src[tid];
    dst[tid + 256] = src[tid + 256];
  }
  __syncthreads();

#pragma unroll 1
  for (int s = 0; s < T_SEQ; ++s) {
    const int t = dir ? (T_SEQ - 1 - s) : s;
    const unsigned short* cur = hn_lds[s & 1];
    unsigned short* nxt = hn_lds[(s & 1) ^ 1];

    v8f zero = {};
    v8f ax_r = zero, ax_z = zero, ax_n = zero;   // input-part (kept separate for n-gate)
    v8f ah_r = zero, ah_z = zero, ah_n = zero;   // hidden-part

    // x-part: hn[t] (16x256) @ w_ih2^T
#pragma unroll
    for (int kt = 0; kt < 8; ++kt) {
      Frag a; load_afrag(a, cur, H1, kt, lane);
      ax_r = WMMA_BF16(a, Bx[0][kt], ax_r);
      ax_z = WMMA_BF16(a, Bx[1][kt], ax_z);
      ax_n = WMMA_BF16(a, Bx[2][kt], ax_n);
    }
    // h-part: h (16x128) @ w_hh2^T
#pragma unroll
    for (int kt = 0; kt < 4; ++kt) {
      Frag a; load_afrag(a, h_bf, H2, kt, lane);
      ah_r = WMMA_BF16(a, Bh[0][kt], ah_r);
      ah_z = WMMA_BF16(a, Bh[1][kt], ah_z);
      ah_n = WMMA_BF16(a, Bh[2][kt], ah_n);
    }

    // overlap: stage next step's hn slice while gate math runs
    if (s + 1 < T_SEQ) {
      const int tn = dir ? (T_SEQ - 2 - s) : (s + 1);
      const uint4* src = (const uint4*)(hn_ws + ((size_t)tn * BATCH + bbase) * H1);
      uint4* dst = (uint4*)nxt;
      dst[tid] = src[tid];
      dst[tid + 256] = src[tid + 256];
    }

    // gates (wave-local); n-gate keeps x/h parts separate — faithful
#pragma unroll
    for (int r = 0; r < 8; ++r) {
      int m = r + (half << 3);
      float rg = sigmoidf_((ax_r[r] + bi_r) + (ah_r[r] + bh_r));
      float zg = sigmoidf_((ax_z[r] + bi_z) + (ah_z[r] + bh_z));
      float ng = tanhf_((ax_n[r] + bi_n) + rg * (ah_n[r] + bh_n));
      float hp = h_f32[m * H2 + j];
      float hnew = (1.0f - zg) * ng + zg * hp;
      h_f32[m * H2 + j] = hnew;
      out[((size_t)t * BATCH + bbase + m) * 256 + dir * 128 + j] = hnew;
    }
    __syncthreads();

    // repack bf16 state (one ds_store_b128 per thread)
    {
      int m = tid >> 4, tx = tid & 15;
      unsigned pw[4];
#pragma unroll
      for (int c = 0; c < 4; ++c) {
        int jj = tx * 8 + 2 * c;
        pw[c] = pk2(h_f32[m * H2 + jj], h_f32[m * H2 + jj + 1]);
      }
      uint4 val; val.x = pw[0]; val.y = pw[1]; val.z = pw[2]; val.w = pw[3];
      *(uint4*)(h_bf + m * H2 + tx * 8) = val;
    }
    __syncthreads();
  }
}

// ---------- host ----------
extern "C" void kernel_launch(void* const* d_in, const int* in_sizes, int n_in,
                              void* d_out, int out_size, void* d_ws, size_t ws_size,
                              hipStream_t stream) {
  const int*   token  = (const int*)  d_in[0];
  const float* emb    = (const float*)d_in[1];
  const float* w_ih1  = (const float*)d_in[2];
  const float* w_hh1  = (const float*)d_in[3];
  const float* b_ih1  = (const float*)d_in[4];
  const float* b_hh1  = (const float*)d_in[5];
  const float* ln_g   = (const float*)d_in[6];
  const float* ln_b   = (const float*)d_in[7];
  const float* w_ih2f = (const float*)d_in[8];
  const float* w_hh2f = (const float*)d_in[9];
  const float* b_ih2f = (const float*)d_in[10];
  const float* b_hh2f = (const float*)d_in[11];
  const float* w_ih2b = (const float*)d_in[12];
  const float* w_hh2b = (const float*)d_in[13];
  const float* b_ih2b = (const float*)d_in[14];
  const float* b_hh2b = (const float*)d_in[15];

  char* ws = (char*)d_ws;
  float*          G1       = (float*)(ws + 0);                 // 128*768 f32   = 393216 B
  unsigned int*   whh1_sw  = (unsigned int*)(ws + 393216);     // 768x256 bf16  = 393216 B
  unsigned int*   wih2f_sw = (unsigned int*)(ws + 786432);     // 384x256 bf16  = 196608 B
  unsigned int*   whh2f_sw = (unsigned int*)(ws + 983040);     // 384x128 bf16  =  98304 B
  unsigned int*   wih2b_sw = (unsigned int*)(ws + 1081344);    // 196608 B
  unsigned int*   whh2b_sw = (unsigned int*)(ws + 1277952);    //  98304 B
  unsigned short* hn_ws    = (unsigned short*)(ws + 1376256);  // 1024*128*256 bf16 = 64 MB

  // Phase 0: fold embedding+bias into a 128x768 gate table; swizzle weights to bf16 tiles
  k_gemb<<<384, 256, 0, stream>>>(emb, w_ih1, b_ih1, G1);
  k_swz <<<384, 256, 0, stream>>>(w_hh1,  whh1_sw,  G1N, H1);
  k_swz <<<192, 256, 0, stream>>>(w_ih2f, wih2f_sw, G2N, H1);
  k_swz <<< 96, 256, 0, stream>>>(w_hh2f, whh2f_sw, G2N, H2);
  k_swz <<<192, 256, 0, stream>>>(w_ih2b, wih2b_sw, G2N, H1);
  k_swz <<< 96, 256, 0, stream>>>(w_hh2b, whh2b_sw, G2N, H2);

  // Phase 1: layer-1 recurrence + LayerNorm (8 batch slices, fully independent)
  k_layer1<<<8, 256, 0, stream>>>(token, G1, whh1_sw, b_hh1, ln_g, ln_b, hn_ws);

  // Phase 2: bidirectional layer-2 recurrence (8 fwd + 8 bwd workgroups)
  k_layer2<<<16, 256, 0, stream>>>(hn_ws, wih2f_sw, whh2f_sw, b_ih2f, b_hh2f,
                                   wih2b_sw, whh2b_sw, b_ih2b, b_hh2b, (float*)d_out);
}
